// WindowAttention_64381559767406
// MI455X (gfx1250) — matrix operand order repaired and checked by reading, not compiled, not run on hardware
//
#include <hip/hip_runtime.h>

// ---------------------------------------------------------------------------
// Fused window attention for MI455X (gfx1250, wave32, WMMA f16 16x16x32).
// One workgroup (8 waves) per window; all intermediates live in LDS.
// ---------------------------------------------------------------------------

typedef __attribute__((ext_vector_type(16))) _Float16 v16h;
typedef __attribute__((ext_vector_type(8)))  float    v8f;
typedef __attribute__((ext_vector_type(4)))  _Float16 v4h;

#define WMMA_F16(A, B, C) \
  __builtin_amdgcn_wmma_f32_16x16x32_f16(false, (A), false, (B), (short)0, (C), false, false)

union V16U { v16h v; float4 f[2]; };

// LDS pitches (halfs): multiples of 8 (16B align); pitch%32dw==4 -> bank spread
constexpr int XP_ = 136;   // x / O f16 [64][128] (+8 pad)
constexpr int QKP = 264;   // Q|K packed [64][256] (+8 pad)
constexpr int VTP = 264;   // V^T packed [32][4*64] (+8 pad)
constexpr int PP  = 72;    // softmax P, per head [64][64] (+8 pad)

// A-matrix fragment (16x32 f16, MxK). CDNA5 layout: lane l holds row (l&15);
// lanes 0-15 hold K {kb+0..7, kb+16..23}, lanes 16-31 hold K {kb+8..15, kb+24..31}.
__device__ inline v16h ldA(const _Float16* base, int pitch, int rowbase, int kb, int lane) {
  const int r   = rowbase + (lane & 15);
  const int sel = (lane & 16) ? 8 : 0;
  const _Float16* p = base + r * pitch + kb + sel;
  V16U u;
  u.f[0] = *reinterpret_cast<const float4*>(p);
  u.f[1] = *reinterpret_cast<const float4*>(p + 16);
  return u.v;
}

// B-matrix fragment (32x16 f16, KxN), read from a row-major N x K buffer
// (i.e. B^T), so each lane reads two 8-wide K runs of its column.
// Lanes 0-15: col n=lane, K {kb+0..7, kb+16..23}; lanes 16-31: col n=lane-16, K {kb+8..15, kb+24..31}.
__device__ inline v16h ldB(const _Float16* base, int pitch, int nbase, int kb, int lane) {
  const int r   = nbase + (lane & 15);
  const int sel = (lane & 16) ? 8 : 0;
  const _Float16* p = base + r * pitch + kb + sel;
  V16U u;
  u.f[0] = *reinterpret_cast<const float4*>(p);
  u.f[1] = *reinterpret_cast<const float4*>(p + 16);
  return u.v;
}

// ---- weight prep: f32 -> f16, transposed (N-major) for B-fragment reads ----
__global__ void prep_weights(const float* __restrict__ Wqkv,
                             const float* __restrict__ Wproj,
                             _Float16* __restrict__ wqkvT,
                             _Float16* __restrict__ wprojT) {
  const int idx = blockIdx.x * blockDim.x + threadIdx.x;   // 0 .. 65535
  if (idx < 384 * 128) {
    const int n = idx >> 7, k = idx & 127;
    wqkvT[idx] = (_Float16)Wqkv[k * 384 + n];
  } else {
    const int j = idx - 384 * 128;
    const int n = j >> 7, k = j & 127;
    wprojT[j] = (_Float16)Wproj[k * 128 + n];
  }
}

// One 64x16 column-strip of the QKV GEMM. T3 (0=Q,1=K,2=V) is a compile-time
// constant so the scatter epilogue is branch-free straight-line code.
template <int T3>
__device__ inline void qkv_strip(const _Float16* __restrict__ sX,
                                 _Float16* __restrict__ sQK,
                                 _Float16* __restrict__ sVT,
                                 const _Float16* __restrict__ wqkvT,
                                 const float* __restrict__ bqkv,
                                 int wave, int lane, int cit, int rAdd) {
  const int nt = T3 * 8 + wave;                 // wave w owns tiles {w, 8+w, 16+w}
  const float bias = bqkv[nt * 16 + cit];
  v16h bw[4];
#pragma unroll
  for (int ks = 0; ks < 4; ++ks) bw[ks] = ldB(wqkvT, 128, nt * 16, ks * 32, lane);
  const int h  = wave >> 1;                     // head of this strip
  const int dd = ((wave & 1) << 4) + cit;       // dim-in-head for this lane
#pragma unroll
  for (int mt = 0; mt < 4; ++mt) {
    v8f acc = {};
#pragma unroll
    for (int ks = 0; ks < 4; ++ks) {
      v16h aX = ldA(sX, XP_, mt * 16, ks * 32, lane);
      acc = WMMA_F16(aX, bw[ks], acc);
    }
#pragma unroll
    for (int i = 0; i < 8; ++i) {
      const int row = mt * 16 + i + rAdd;
      const _Float16 hv = (_Float16)(acc[i] + bias);
      if (T3 == 0)      sQK[row * QKP + h * 32 + dd]       = hv;  // Q
      else if (T3 == 1) sQK[row * QKP + 128 + h * 32 + dd] = hv;  // K
      else              sVT[dd * VTP + h * 64 + row]       = hv;  // V^T
    }
  }
}

// ---- fused kernel -----------------------------------------------------------
__global__ __launch_bounds__(256) void winattn_fused(
    const float*    __restrict__ x,       // [4096,49,128]
    const _Float16* __restrict__ wqkvT,   // [384,128]
    const _Float16* __restrict__ wprojT,  // [128,128]
    const float*    __restrict__ bqkv,    // [384]
    const float*    __restrict__ bproj,   // [128]
    float*          __restrict__ out)     // [4096,49,128]
{
  constexpr float SCALE = 0.08838834764831845f;   // 128^-0.5

  extern __shared__ _Float16 smem[];
  _Float16* sX  = smem;                // 8704 halfs used of 9216-half region
  _Float16* sP  = smem;                // alias: P for 2 heads = 2*64*72 = 9216 halfs
  _Float16* sQK = smem + 9216;         // 64*264 = 16896 halfs
  _Float16* sVT = sQK + 16896;         // 32*264 = 8448 halfs
  _Float16* sO  = sVT + 8448;          // 64*136 = 8704 halfs
  // total 43264 halfs = 86528 bytes

  const int tid  = threadIdx.x;
  const int lane = tid & 31;
  const int wave = tid >> 5;
  const int cit  = lane & 15;               // column within 16-wide tile
  const int rAdd = (lane & 16) ? 8 : 0;     // C/D accumulator row offset
  const int b    = blockIdx.x;
  const float* xb = x + (size_t)b * (49 * 128);

  // ---------------- Phase A: load x[b] -> f16 LDS, zero-pad rows 49..63 ----
  for (int idx = tid; idx < 49 * 32; idx += 256) {
    const int row = idx >> 5, c4 = (idx & 31) << 2;
    const float4 xv = reinterpret_cast<const float4*>(xb)[idx];
    v4h h4 = { (_Float16)xv.x, (_Float16)xv.y, (_Float16)xv.z, (_Float16)xv.w };
    *reinterpret_cast<v4h*>(&sX[row * XP_ + c4]) = h4;
  }
  for (int idx = tid; idx < 15 * 32; idx += 256) {
    const int row = 49 + (idx >> 5), c4 = (idx & 31) << 2;
    v4h z = {};
    *reinterpret_cast<v4h*>(&sX[row * XP_ + c4]) = z;
  }
  __syncthreads();

  // ---------------- Phase B: QKV = X(64x128) @ Wqkv(128x384) + bias --------
  qkv_strip<0>(sX, sQK, sVT, wqkvT, bqkv, wave, lane, cit, rAdd);  // Q
  qkv_strip<1>(sX, sQK, sVT, wqkvT, bqkv, wave, lane, cit, rAdd);  // K
  qkv_strip<2>(sX, sQK, sVT, wqkvT, bqkv, wave, lane, cit, rAdd);  // V
  __syncthreads();

  // ---------------- Phases C/D: attention, two heads at a time -------------
#pragma unroll
  for (int hp = 0; hp < 2; ++hp) {
    { // --- C: S = Q K^T * scale, softmax, P -> LDS (f16) ---
      const int lh = wave >> 2;            // local head 0/1
      const int h  = hp * 2 + lh;
      const int mt = wave & 3;             // row strip
      const v16h aq = ldA(sQK + h * 32, QKP, mt * 16, 0, lane);   // K-dim = d = 32
      v8f s[4];
#pragma unroll
      for (int nt = 0; nt < 4; ++nt) {
        v16h bk = ldB(sQK + 128 + h * 32, QKP, nt * 16, 0, lane);
        v8f z = {};
        s[nt] = WMMA_F16(aq, bk, z);
      }
#pragma unroll
      for (int nt = 0; nt < 4; ++nt) {
        const bool oob = (nt * 16 + cit) >= 49;
#pragma unroll
        for (int i = 0; i < 8; ++i)
          s[nt][i] = oob ? -3.0e38f : s[nt][i] * SCALE;
      }
      _Float16* Pl = sP + lh * (64 * PP);
#pragma unroll
      for (int i = 0; i < 8; ++i) {        // row (mt*16+i+rAdd) lives in a 16-lane half
        float m = fmaxf(fmaxf(s[0][i], s[1][i]), fmaxf(s[2][i], s[3][i]));
        m = fmaxf(m, __shfl_xor(m, 8));
        m = fmaxf(m, __shfl_xor(m, 4));
        m = fmaxf(m, __shfl_xor(m, 2));
        m = fmaxf(m, __shfl_xor(m, 1));
        float sum = 0.f;
#pragma unroll
        for (int nt = 0; nt < 4; ++nt) {
          const float p = __expf(s[nt][i] - m);
          s[nt][i] = p;
          sum += p;
        }
        sum += __shfl_xor(sum, 8);
        sum += __shfl_xor(sum, 4);
        sum += __shfl_xor(sum, 2);
        sum += __shfl_xor(sum, 1);
        const float inv = 1.0f / sum;
        const int row = mt * 16 + i + rAdd;
#pragma unroll
        for (int nt = 0; nt < 4; ++nt)
          Pl[row * PP + nt * 16 + cit] = (_Float16)(s[nt][i] * inv);
      }
    }
    __syncthreads();
    { // --- D: O = P(64x64) @ V(64x32) per head ---
#pragma unroll
      for (int rep = 0; rep < 2; ++rep) {
        const int t  = wave + rep * 8;     // 16 tiles: 2 heads x 4 mt x 2 nt
        const int lh = t >> 3;
        const int mt = (t & 7) >> 1;
        const int nt = t & 1;
        const int h  = hp * 2 + lh;
        v8f o = {};
#pragma unroll
        for (int ks = 0; ks < 2; ++ks) {
          v16h ap = ldA(sP + lh * (64 * PP), PP, mt * 16, ks * 32, lane);
          v16h bv = ldB(sVT + h * 64, VTP, nt * 16, ks * 32, lane);
          o = WMMA_F16(ap, bv, o);
        }
        const int cb = h * 32 + nt * 16 + cit;
#pragma unroll
        for (int i = 0; i < 8; ++i)
          sO[(mt * 16 + i + rAdd) * XP_ + cb] = (_Float16)o[i];
      }
    }
    __syncthreads();
  }

  // ---------------- Phase E: out = O(64x128) @ Wproj(128x128) + bias -------
  {
    const size_t ob = (size_t)b * (49 * 128);
#pragma unroll
    for (int j = 0; j < 4; ++j) {
      const int t  = j * 8 + wave;         // 32 tiles: 4 mt x 8 nt
      const int mt = t >> 3;
      const int nt = t & 7;
      const float bias = bproj[nt * 16 + cit];
      v8f acc = {};
#pragma unroll
      for (int ks = 0; ks < 4; ++ks) {
        v16h aO = ldA(sO, XP_, mt * 16, ks * 32, lane);
        v16h bw = ldB(wprojT, 128, nt * 16, ks * 32, lane);
        acc = WMMA_F16(aO, bw, acc);
      }
#pragma unroll
      for (int i = 0; i < 8; ++i) {
        const int row = mt * 16 + i + rAdd;
        if (row < 49)
          out[ob + row * 128 + nt * 16 + cit] = acc[i] + bias;
      }
    }
  }
}

// ---------------------------------------------------------------------------
extern "C" void kernel_launch(void* const* d_in, const int* in_sizes, int n_in,
                              void* d_out, int out_size, void* d_ws, size_t ws_size,
                              hipStream_t stream) {
  (void)in_sizes; (void)n_in; (void)out_size; (void)ws_size;
  const float* x     = (const float*)d_in[0];
  const float* Wqkv  = (const float*)d_in[1];
  const float* bqkv  = (const float*)d_in[2];
  const float* Wproj = (const float*)d_in[3];
  const float* bproj = (const float*)d_in[4];
  float* out = (float*)d_out;

  _Float16* wqkvT  = (_Float16*)d_ws;            // 384*128 f16 = 96 KB
  _Float16* wprojT = wqkvT + 384 * 128;          // 128*128 f16 = 32 KB

  prep_weights<<<256, 256, 0, stream>>>(Wqkv, Wproj, wqkvT, wprojT);

  const int smemBytes = (9216 + 16896 + 8448 + 8704) * 2;   // 86528 B dynamic LDS
  winattn_fused<<<4096, 256, smemBytes, stream>>>(x, wqkvT, wprojT, bqkv, bproj, out);
}
